// DecoderActor_89404039233955
// MI455X (gfx1250) — matrix operand order repaired and checked by reading, not compiled, not run on hardware
//
#include <hip/hip_runtime.h>
#include <hip/hip_bf16.h>

// ---------------------------------------------------------------------------
// DecoderActor for MI455X (gfx1250, wave32, WMMA bf16 16x16x32, f32 accum)
// ---------------------------------------------------------------------------

typedef __attribute__((ext_vector_type(16))) __bf16 v16bf;
typedef __attribute__((ext_vector_type(8)))  float  v8f;

#define BSZ 1024
#define NN  128
#define TT  16
#define EE  128
#define FF  64
#define ES  2048

static __device__ __forceinline__ v8f wmma_bf16(v16bf a, v16bf b, v8f c) {
    return __builtin_amdgcn_wmma_f32_16x16x32_bf16(false, a, false, b,
                                                   (short)0, c, false, false);
}

static __device__ __forceinline__ float sigmoidf_(float x) {
    return 1.0f / (1.0f + __expf(-x));
}

// ---------------------------------------------------------------------------
// k_prep: convert GRU weights to bf16; transpose gcn_w [F,E] -> [E,F] bf16
// ---------------------------------------------------------------------------
__global__ void k_prep(const float* __restrict__ ewih, const float* __restrict__ ewhh,
                       const float* __restrict__ dwhh, const float* __restrict__ gw,
                       __bf16* __restrict__ ewihb, __bf16* __restrict__ ewhhb,
                       __bf16* __restrict__ dwhhb, __bf16* __restrict__ gwtb) {
    int i = blockIdx.x * blockDim.x + threadIdx.x;
    if (i < 3 * EE * EE) {   // 49152
        ewihb[i] = (__bf16)ewih[i];
        ewhhb[i] = (__bf16)ewhh[i];
        dwhhb[i] = (__bf16)dwhh[i];
    }
    if (i < EE * FF) {       // 8192, gwt[e*64+f] = gw[f*128+e]
        int e = i >> 6, f = i & 63;
        gwtb[i] = (__bf16)gw[f * EE + e];
    }
}

// ---------------------------------------------------------------------------
// k_adj: deterministic dense normalized adjacency (self-loops + sym norm)
// thread (d,s): A[d,s] = count(edges d<-s)*dinv[s]*dinv[d] (+dinv^2 if d==s)
// ---------------------------------------------------------------------------
__global__ void k_adj(const int* __restrict__ edge_index, __bf16* __restrict__ A_bf) {
    int i = blockIdx.x * blockDim.x + threadIdx.x;   // 16384 threads
    if (i >= NN * NN) return;
    int d = i >> 7, s = i & 127;
    const int* src = edge_index;
    const int* dst = edge_index + ES;
    int cnt = 0, degd = 1, degs = 1;                 // self-loop included
    for (int e = 0; e < ES; ++e) {
        int es = src[e], ed = dst[e];
        cnt  += (ed == d && es == s);
        degd += (ed == d);
        degs += (ed == s);
    }
    float dinv_d = rsqrtf((float)degd);
    float dinv_s = rsqrtf((float)degs);
    float v = (float)cnt * dinv_d * dinv_s;
    if (d == s) v += dinv_d * dinv_d;
    A_bf[i] = (__bf16)v;
}

// ---------------------------------------------------------------------------
// k_dec_gi: decoder input gates, shared across batch.
// dec_gi[t*384+o] = dec_bih[o] + node_embed[t,:]·dec_wih[o,:]
// ---------------------------------------------------------------------------
__global__ void k_dec_gi(const float* __restrict__ dwih, const float* __restrict__ dbih,
                         const float* __restrict__ nemb, float* __restrict__ dec_gi) {
    int i = blockIdx.x * blockDim.x + threadIdx.x;
    if (i >= TT * 3 * EE) return;
    int t = i / (3 * EE), o = i % (3 * EE);
    const float* x = nemb + t * EE;
    const float* w = dwih + o * EE;
    float s = dbih[o];
#pragma unroll 8
    for (int e = 0; e < EE; ++e) s += w[e] * x[e];
    dec_gi[i] = s;
}

// ---------------------------------------------------------------------------
// k_gcn: one block per batch row b.
//   GEMM1: h = net_x[b] @ gcn_wT.T  -> LDS transposed as hT[e][m] (bf16)
//   GEMM2: M = A @ h ; ns_dot[b,n] = sum_e relu(M[n,e]+gcn_b[e])*mlp_w[e]
// ---------------------------------------------------------------------------
__global__ __launch_bounds__(256) void k_gcn(const float* __restrict__ net_x,
                                             const __bf16* __restrict__ gwt,
                                             const float* __restrict__ gcn_b,
                                             const float* __restrict__ mlp_w,
                                             const __bf16* __restrict__ A_bf,
                                             float* __restrict__ ns_dot) {
    __shared__ __bf16 hT[NN * NN];   // 32 KB, hT[e*128+m]
    int b = blockIdx.x;
    int tid = threadIdx.x;
    int w = tid >> 5, lane = tid & 31;
    int row = lane & 15, hi = lane >> 4;

    // ---- GEMM1: m-tile = w, loop e-tiles; A-frags (net_x rows) loaded once
    const float* X = net_x + (size_t)b * NN * FF;
    v16bf a0, a1;
    {
        const float* p = X + (16 * w + row) * FF + hi * 16;
#pragma unroll
        for (int j = 0; j < 8; ++j) {
            a0[j]     = (__bf16)p[j];
            a0[8 + j] = (__bf16)p[8 + j];
            a1[j]     = (__bf16)p[32 + j];
            a1[8 + j] = (__bf16)p[40 + j];
        }
    }
#pragma unroll
    for (int et = 0; et < 8; ++et) {
        v16bf b0, b1;
        const __bf16* q = gwt + (et * 16 + row) * FF + hi * 16;
#pragma unroll
        for (int i = 0; i < 16; ++i) { b0[i] = q[i]; b1[i] = q[32 + i]; }
        v8f acc = {0.f, 0.f, 0.f, 0.f, 0.f, 0.f, 0.f, 0.f};
        acc = wmma_bf16(a0, b0, acc);
        acc = wmma_bf16(a1, b1, acc);
        __bf16* d = hT + (et * 16 + row) * NN + 16 * w + 8 * hi;
#pragma unroll
        for (int r = 0; r < 8; ++r) d[r] = (__bf16)acc[r];
    }
    __syncthreads();

    // ---- GEMM2: n-tile = w; A-frags (adjacency rows) loaded once, K=128
    v16bf A0, A1, A2, A3;
    {
        const __bf16* p = A_bf + (16 * w + row) * NN + hi * 16;
#pragma unroll
        for (int j = 0; j < 8; ++j) {
            A0[j] = p[j];      A0[8 + j] = p[8 + j];
            A1[j] = p[32 + j]; A1[8 + j] = p[40 + j];
            A2[j] = p[64 + j]; A2[8 + j] = p[72 + j];
            A3[j] = p[96 + j]; A3[8 + j] = p[104 + j];
        }
    }
    float partial[8] = {0.f, 0.f, 0.f, 0.f, 0.f, 0.f, 0.f, 0.f};
#pragma unroll
    for (int et = 0; et < 8; ++et) {
        v8f acc = {0.f, 0.f, 0.f, 0.f, 0.f, 0.f, 0.f, 0.f};
        const __bf16* q = hT + (et * 16 + row) * NN + hi * 16;
        v16bf bb;
#pragma unroll
        for (int i = 0; i < 16; ++i) bb[i] = q[i];
        acc = wmma_bf16(A0, bb, acc);
#pragma unroll
        for (int i = 0; i < 16; ++i) bb[i] = q[32 + i];
        acc = wmma_bf16(A1, bb, acc);
#pragma unroll
        for (int i = 0; i < 16; ++i) bb[i] = q[64 + i];
        acc = wmma_bf16(A2, bb, acc);
#pragma unroll
        for (int i = 0; i < 16; ++i) bb[i] = q[96 + i];
        acc = wmma_bf16(A3, bb, acc);

        int e = et * 16 + row;
        float gb = gcn_b[e], mw = mlp_w[e];
#pragma unroll
        for (int r = 0; r < 8; ++r) {
            float v = acc[r] + gb;
            v = v > 0.f ? v : 0.f;
            partial[r] += v * mw;
        }
    }
    // reduce over 16 lanes (each half-wave owns one 8-row n-group)
#pragma unroll
    for (int off = 1; off < 16; off <<= 1)
#pragma unroll
        for (int r = 0; r < 8; ++r)
            partial[r] += __shfl_xor(partial[r], off, 32);
    if (row == 0) {
#pragma unroll
        for (int r = 0; r < 8; ++r)
            ns_dot[(size_t)b * NN + 16 * w + 8 * hi + r] = partial[r];
    }
}

// ---------------------------------------------------------------------------
// k_rnn: persistent encoder (18 steps) + decoder (16 steps).
// One block = 16 batch rows, 8 waves; h kept in LDS. Each step:
// gates = [x@wih.T] + [h@whh.T] via WMMA bf16, then elementwise GRU math.
// Decoder emits hdot[b,t] = hn·mlp_w (logits are rank-1: ns_dot+hdot+mlp_b).
// ---------------------------------------------------------------------------
__global__ __launch_bounds__(256) void k_rnn(
        const float* __restrict__ sfc, const int* __restrict__ sdest,
        const float* __restrict__ nemb,
        const __bf16* __restrict__ ewih, const __bf16* __restrict__ ewhh,
        const float* __restrict__ ebih, const float* __restrict__ ebhh,
        const __bf16* __restrict__ dwhh, const float* __restrict__ dbhh,
        const float* __restrict__ dec_gi, const float* __restrict__ mlp_w,
        float* __restrict__ hdot) {
    __shared__ float  h32[16][EE];      //  8 KB
    __shared__ __bf16 hbf[16][EE];      //  4 KB
    __shared__ __bf16 xbf[16][EE];      //  4 KB
    __shared__ float  gsum[16][2 * EE]; // 16 KB (r,z pre-activations)
    __shared__ float  gin[16][EE];      //  8 KB (input n-gate)
    __shared__ float  ghn[16][EE];      //  8 KB (hidden n-gate)

    int tid = threadIdx.x;
    int w = tid >> 5, lane = tid & 31;
    int lrow = lane & 15, hi = lane >> 4;
    int b0 = blockIdx.x * 16;
    int erow = tid >> 4, ecol = (tid & 15) * 8;

#pragma unroll
    for (int j = 0; j < 8; ++j) {
        h32[erow][ecol + j] = 0.f;
        hbf[erow][ecol + j] = (__bf16)0.f;
    }
    __syncthreads();

    // column bases of this wave's three gate tiles
    const int cr = 16 * w, cz = EE + 16 * w, cn = 2 * EE + 16 * w;

    // ======================= ENCODER: 18 steps =======================
    for (int t = 0; t < 18; ++t) {
        if (t < 16) {
            const float* p = sfc + (((size_t)(b0 + erow)) * TT + t) * EE + ecol;
#pragma unroll
            for (int j = 0; j < 8; ++j) xbf[erow][ecol + j] = (__bf16)p[j];
        } else {
            int idx = sdest[(b0 + erow) * 2 + (t - 16)];
            const float* p = nemb + (size_t)idx * EE + ecol;
#pragma unroll
            for (int j = 0; j < 8; ++j) xbf[erow][ecol + j] = (__bf16)p[j];
        }
        __syncthreads();

        v8f ar = {0.f,0.f,0.f,0.f,0.f,0.f,0.f,0.f};
        v8f az = ar, ain = ar, ahn = ar;
#pragma unroll
        for (int k = 0; k < 4; ++k) {
            int k0 = k * 32;
            v16bf ax, ah;
#pragma unroll
            for (int j = 0; j < 8; ++j) {
                ax[j]     = xbf[lrow][k0 + hi * 16 + j];
                ax[8 + j] = xbf[lrow][k0 + 8 + hi * 16 + j];
                ah[j]     = hbf[lrow][k0 + hi * 16 + j];
                ah[8 + j] = hbf[lrow][k0 + 8 + hi * 16 + j];
            }
            v16bf bri, brh, bzi, bzh, bni, bnh;
            const __bf16* pri = ewih + (cr + lrow) * EE + k0 + hi * 16;
            const __bf16* prh = ewhh + (cr + lrow) * EE + k0 + hi * 16;
            const __bf16* pzi = ewih + (cz + lrow) * EE + k0 + hi * 16;
            const __bf16* pzh = ewhh + (cz + lrow) * EE + k0 + hi * 16;
            const __bf16* pni = ewih + (cn + lrow) * EE + k0 + hi * 16;
            const __bf16* pnh = ewhh + (cn + lrow) * EE + k0 + hi * 16;
#pragma unroll
            for (int i = 0; i < 16; ++i) {
                bri[i] = pri[i]; brh[i] = prh[i];
                bzi[i] = pzi[i]; bzh[i] = pzh[i];
                bni[i] = pni[i]; bnh[i] = pnh[i];
            }
            ar  = wmma_bf16(ax, bri, ar);  ar  = wmma_bf16(ah, brh, ar);
            az  = wmma_bf16(ax, bzi, az);  az  = wmma_bf16(ah, bzh, az);
            ain = wmma_bf16(ax, bni, ain); ahn = wmma_bf16(ah, bnh, ahn);
        }
#pragma unroll
        for (int r = 0; r < 8; ++r) {
            int m = r + 8 * hi;
            gsum[m][cr + lrow]      = ar[r];
            gsum[m][EE + cr + lrow] = az[r];
            gin[m][cr + lrow]       = ain[r];
            ghn[m][cr + lrow]       = ahn[r];
        }
        __syncthreads();

#pragma unroll
        for (int j = 0; j < 8; ++j) {
            int e = ecol + j;
            float rg = sigmoidf_(gsum[erow][e]      + ebih[e]      + ebhh[e]);
            float zg = sigmoidf_(gsum[erow][EE + e] + ebih[EE + e] + ebhh[EE + e]);
            float ng = tanhf(gin[erow][e] + ebih[2 * EE + e] +
                             rg * (ghn[erow][e] + ebhh[2 * EE + e]));
            float hn = (1.f - zg) * ng + zg * h32[erow][e];
            h32[erow][e] = hn;
            hbf[erow][e] = (__bf16)hn;
        }
        __syncthreads();
    }

    // ======================= DECODER: 16 steps =======================
    for (int t = 0; t < 16; ++t) {
        v8f ar = {0.f,0.f,0.f,0.f,0.f,0.f,0.f,0.f};
        v8f az = ar, ahn = ar;
#pragma unroll
        for (int k = 0; k < 4; ++k) {
            int k0 = k * 32;
            v16bf ah;
#pragma unroll
            for (int j = 0; j < 8; ++j) {
                ah[j]     = hbf[lrow][k0 + hi * 16 + j];
                ah[8 + j] = hbf[lrow][k0 + 8 + hi * 16 + j];
            }
            v16bf brh, bzh, bnh;
            const __bf16* prh = dwhh + (cr + lrow) * EE + k0 + hi * 16;
            const __bf16* pzh = dwhh + (cz + lrow) * EE + k0 + hi * 16;
            const __bf16* pnh = dwhh + (cn + lrow) * EE + k0 + hi * 16;
#pragma unroll
            for (int i = 0; i < 16; ++i) {
                brh[i] = prh[i]; bzh[i] = pzh[i]; bnh[i] = pnh[i];
            }
            ar  = wmma_bf16(ah, brh, ar);
            az  = wmma_bf16(ah, bzh, az);
            ahn = wmma_bf16(ah, bnh, ahn);
        }
#pragma unroll
        for (int r = 0; r < 8; ++r) {
            int m = r + 8 * hi;
            gsum[m][cr + lrow]      = ar[r];
            gsum[m][EE + cr + lrow] = az[r];
            ghn[m][cr + lrow]       = ahn[r];
        }
        __syncthreads();

        const float* gi = dec_gi + t * 3 * EE;
        float part = 0.f;
#pragma unroll
        for (int j = 0; j < 8; ++j) {
            int e = ecol + j;
            float rg = sigmoidf_(gi[e]      + gsum[erow][e]      + dbhh[e]);
            float zg = sigmoidf_(gi[EE + e] + gsum[erow][EE + e] + dbhh[EE + e]);
            float ng = tanhf(gi[2 * EE + e] + rg * (ghn[erow][e] + dbhh[2 * EE + e]));
            float hn = (1.f - zg) * ng + zg * h32[erow][e];
            h32[erow][e] = hn;
            hbf[erow][e] = (__bf16)hn;
            part += hn * mlp_w[e];
        }
        // 16 threads per batch row are 16 consecutive lanes -> shfl reduce
#pragma unroll
        for (int off = 1; off < 16; off <<= 1)
            part += __shfl_xor(part, off, 32);
        if ((tid & 15) == 0) hdot[(size_t)(b0 + erow) * TT + t] = part;
        __syncthreads();
    }
}

// ---------------------------------------------------------------------------
// k_final: logits[b,t,n] = ns_dot[b,n] + hdot[b,t] + mlp_b;
// softmax over n is t-invariant (additive shift) = softmax(ns_dot[b,:]).
// ---------------------------------------------------------------------------
__global__ __launch_bounds__(128) void k_final(const float* __restrict__ ns_dot,
                                               const float* __restrict__ hdot,
                                               const float* __restrict__ mlp_b,
                                               float* __restrict__ out) {
    __shared__ float red[8];
    int b = blockIdx.x, n = threadIdx.x;
    float v = ns_dot[(size_t)b * NN + n];

    float m = v;
#pragma unroll
    for (int off = 1; off < 32; off <<= 1) m = fmaxf(m, __shfl_xor(m, off, 32));
    if ((n & 31) == 0) red[n >> 5] = m;
    __syncthreads();
    m = fmaxf(fmaxf(red[0], red[1]), fmaxf(red[2], red[3]));

    float ex = __expf(v - m);
    float s = ex;
#pragma unroll
    for (int off = 1; off < 32; off <<= 1) s += __shfl_xor(s, off, 32);
    if ((n & 31) == 0) red[4 + (n >> 5)] = s;
    __syncthreads();
    s = red[4] + red[5] + red[6] + red[7];
    float soft = ex / s;

    float mb = mlp_b[0];
    const float* hd = hdot + (size_t)b * TT;
    float* o1 = out + (size_t)b * TT * NN + n;
    float* o2 = o1 + (size_t)BSZ * TT * NN;
#pragma unroll
    for (int t = 0; t < 16; ++t) {
        o1[t * NN] = v + hd[t] + mb;
        o2[t * NN] = soft;
    }
}

// ---------------------------------------------------------------------------
extern "C" void kernel_launch(void* const* d_in, const int* in_sizes, int n_in,
                              void* d_out, int out_size, void* d_ws, size_t ws_size,
                              hipStream_t stream) {
    (void)in_sizes; (void)n_in; (void)out_size; (void)ws_size;
    const float* net_x = (const float*)d_in[0];
    const float* sfc   = (const float*)d_in[1];
    const int*   edge  = (const int*)d_in[2];
    const int*   sdest = (const int*)d_in[3];
    const float* nemb  = (const float*)d_in[4];
    const float* gcn_w = (const float*)d_in[5];
    const float* gcn_b = (const float*)d_in[6];
    const float* ewih  = (const float*)d_in[7];
    const float* ewhh  = (const float*)d_in[8];
    const float* ebih  = (const float*)d_in[9];
    const float* ebhh  = (const float*)d_in[10];
    const float* dwih  = (const float*)d_in[11];
    const float* dwhh  = (const float*)d_in[12];
    const float* dbih  = (const float*)d_in[13];
    const float* dbhh  = (const float*)d_in[14];
    const float* mlpw  = (const float*)d_in[15];
    const float* mlpb  = (const float*)d_in[16];
    float* out = (float*)d_out;

    // workspace layout (all offsets multiple of 256B), total < 1 MB
    char* wsb = (char*)d_ws;
    __bf16* A_bf   = (__bf16*)(wsb + 0);        //  32768  [128,128]
    __bf16* gwt    = (__bf16*)(wsb + 32768);    //  16384  [128,64]
    __bf16* ewihb  = (__bf16*)(wsb + 49152);    //  98304  [384,128]
    __bf16* ewhhb  = (__bf16*)(wsb + 147456);   //  98304
    __bf16* dwhhb  = (__bf16*)(wsb + 245760);   //  98304
    float*  dec_gi = (float*) (wsb + 344064);   //  24576  [16,384]
    float*  ns_dot = (float*) (wsb + 368640);   // 524288  [1024,128]
    float*  hdot   = (float*) (wsb + 892928);   //  65536  [1024,16]

    k_prep  <<<192,  256, 0, stream>>>(ewih, ewhh, dwhh, gcn_w, ewihb, ewhhb, dwhhb, gwt);
    k_adj   <<<64,   256, 0, stream>>>(edge, A_bf);
    k_dec_gi<<<24,   256, 0, stream>>>(dwih, dbih, nemb, dec_gi);
    k_gcn   <<<BSZ,  256, 0, stream>>>(net_x, gwt, gcn_b, mlpw, A_bf, ns_dot);
    k_rnn   <<<BSZ/16, 256, 0, stream>>>(sfc, sdest, nemb, ewihb, ewhhb, ebih, ebhh,
                                         dwhhb, dbhh, dec_gi, mlpw, hdot);
    k_final <<<BSZ,  128, 0, stream>>>(ns_dot, hdot, mlpb, out);
}
